// Encoder_35124242547150
// MI455X (gfx1250) — matrix operand order repaired and checked by reading, not compiled, not run on hardware
//
#include <hip/hip_runtime.h>
#include <hip/hip_bf16.h>

typedef __attribute__((ext_vector_type(16))) __bf16 v16bf;
typedef __attribute__((ext_vector_type(8)))  float  v8f;

#define B_   256
#define T_   1024
#define E_   32
#define H_   128
#define NOBS 1048576

// ---------------------------------------------------------------------------
// Hardware transcendental tanh (gfx1250 TRANS pipe); sigmoid via tanh.
// ---------------------------------------------------------------------------
__device__ __forceinline__ float htanh(float x) {
#if __has_builtin(__builtin_amdgcn_tanhf)
  return __builtin_amdgcn_tanhf(x);
#else
  float r;
  asm volatile("v_tanh_f32 %0, %1\n\tv_nop" : "=v"(r) : "v"(x));
  return r;
#endif
}
__device__ __forceinline__ float hsigmoid(float x) {
  return __builtin_fmaf(0.5f, htanh(0.5f * x), 0.5f);
}

// ---------------------------------------------------------------------------
// Kernel 1: fused ragged embedding-bag + action-embedding select.
// One wave (32 lanes == E) per slot s = b*T + t. obs_slot is sorted, so each
// wave binary-searches its segment bounds and segment-sums obs_emb rows.
// Deterministic (no float atomics). Writes the f32 `embedded` output directly.
// ---------------------------------------------------------------------------
__global__ __launch_bounds__(256) void embed_kernel(
    const int*   __restrict__ obs_ids,
    const int*   __restrict__ obs_slot,
    const int*   __restrict__ action_ids,
    const int*   __restrict__ is_action,
    const float* __restrict__ action_emb,
    const float* __restrict__ obs_emb,
    float*       __restrict__ embedded)   // [B*T, E]
{
  const int lane = threadIdx.x & 31;
  const int slot = (int)((blockIdx.x * (unsigned)blockDim.x + threadIdx.x) >> 5);
  if (slot >= B_ * T_) return;

  float v;
  if (is_action[slot]) {
    v = action_emb[action_ids[slot] * E_ + lane];
  } else {
    int lo = 0, hi = NOBS;
    while (lo < hi) { int mid = (lo + hi) >> 1; if (obs_slot[mid] < slot) lo = mid + 1; else hi = mid; }
    int l2 = lo, h2 = NOBS;
    while (l2 < h2) { int mid = (l2 + h2) >> 1; if (obs_slot[mid] <= slot) l2 = mid + 1; else h2 = mid; }
    float acc = 0.f;
    for (int j = lo; j < l2; ++j)
      acc += obs_emb[obs_ids[j] * E_ + lane];
    v = acc;
  }
  embedded[(size_t)slot * E_ + lane] = v;
}

// ---------------------------------------------------------------------------
// Kernel 2: persistent LSTM, gate-aligned tiling, ONE barrier per step.
// 16 blocks x 256 threads (8 waves). Wave w owns hidden units [w*16, w*16+16):
// its four WMMA N-tiles sit at gate columns q*128 + w*16 (q = i,f,g,o), so all
// four gates of unit j land in the same lane / accumulator element and the
// LSTM cell runs on WMMA accumulators in registers. Bias lives in persistent
// v8f C-vectors consumed as WMMA src2 (dst != src2) -> zero per-step acc-init
// movs. lds_x / lds_h are double-buffered so the elementwise h writeback never
// conflicts with other waves' A-fragment reads: single __syncthreads per step.
// ---------------------------------------------------------------------------
#define XS 40            // padded row stride (bf16) for x tile : banks staggered
#define HS 136           // padded row stride (bf16) for h tile : 16 rows -> 16 bank groups
#define XBUF (16 * XS)
#define HBUF (16 * HS)
__global__ __launch_bounds__(256) void lstm_kernel(
    const float* __restrict__ emb,      // [B,T,E] f32 (embedded section of d_out)
    const int*   __restrict__ lengths,  // [B]
    const float* __restrict__ W_ih,     // [4H,E]
    const float* __restrict__ W_hh,     // [4H,H]
    const float* __restrict__ b_ih,     // [4H]
    const float* __restrict__ b_hh,     // [4H]
    float* __restrict__ outs,           // [B,T,H]
    float* __restrict__ h_out,          // [B,H]
    float* __restrict__ c_out)          // [B,H]
{
  __shared__ __bf16 lds_x[2 * XBUF];
  __shared__ __bf16 lds_h[2 * HBUF];

  const int tid  = threadIdx.x;
  const int lane = tid & 31;
  const int wv   = tid >> 5;            // wave 0..7 -> hidden units [wv*16, wv*16+16)
  const int b0   = blockIdx.x * 16;

  // B-fragment lane mapping (16x16x32 bf16, B = 32x16 KxN):
  //   col = lane&15 ; lane<16 covers K {0..7,16..23}, lane>=16 covers K {8..15,24..31}
  const int colB   = lane & 15;
  const int hiHalf = lane >> 4;         // 0 or 1
  const int j      = wv * 16 + colB;    // hidden unit this lane owns

  // ---- one-time weight fragment build (f32 global -> bf16 registers) ----
  v16bf Bh[4][4];                        // [gate q][k-chunk]
  v16bf Bx[4];                           // [gate q], K=32 -> 1 chunk
  v8f   biasv[4];                        // persistent C-input: bias broadcast
  #pragma unroll
  for (int q = 0; q < 4; ++q) {
    const int n = q * 128 + j;                     // gate column (B[k][n] = W[n][k])
    const float bq = b_ih[n] + b_hh[n];
    #pragma unroll
    for (int r = 0; r < 8; ++r) biasv[q][r] = bq;
    const float* sx = W_ih + n * E_ + hiHalf * 8;
    #pragma unroll
    for (int e = 0; e < 8; ++e) { Bx[q][e] = (__bf16)sx[e]; Bx[q][e + 8] = (__bf16)sx[16 + e]; }
    #pragma unroll
    for (int kc = 0; kc < 4; ++kc) {
      const float* sh = W_hh + n * H_ + kc * 32 + hiHalf * 8;
      #pragma unroll
      for (int e = 0; e < 8; ++e) { Bh[q][kc][e] = (__bf16)sh[e]; Bh[q][kc][e + 8] = (__bf16)sh[16 + e]; }
    }
  }

  // C/D layout: element r <-> batch row m = r + hiHalf*8, col = lane&15 (== j)
  int len8[8];
  #pragma unroll
  for (int r = 0; r < 8; ++r) len8[r] = lengths[b0 + r + hiHalf * 8];

  float c_st[8], h_st[8];
  #pragma unroll
  for (int r = 0; r < 8; ++r) { c_st[r] = 0.f; h_st[r] = 0.f; }
  for (int i = tid; i < 2 * HBUF; i += 256) lds_h[i] = (__bf16)0.f;

  // A-fragment lane mapping (16x32 bf16): row = lane&15;
  // lane<16: K {0..7,16..23}; lane>=16: K {8..15,24..31}
  const int arow = lane & 15;
  const int klo  = hiHalf * 8;

  // ---- x staging: 2 elements/thread; double-buffered in registers ----
  const int xm  = tid >> 4;              // 0..15
  const int xe2 = (tid & 15) * 2;        // 0..30
  const float* xp = emb + ((size_t)(b0 + xm) * T_) * E_ + xe2;
  float xv0 = xp[0], xv1 = xp[1];

  __syncthreads();

  for (int t = 0; t < T_; ++t) {
    const int cur = t & 1;               // write buffer (x(t), h(t))
    const int prv = cur ^ 1;             // read buffer for h(t-1)

    // stage x(t) from registers; immediately issue load of x(t+1)
    __bf16* xw = lds_x + cur * XBUF;
    xw[xm * XS + xe2]     = (__bf16)xv0;
    xw[xm * XS + xe2 + 1] = (__bf16)xv1;
    if (t + 1 < T_) {
      const float* nx = xp + (size_t)(t + 1) * E_;
      xv0 = nx[0]; xv1 = nx[1];
      __builtin_prefetch(nx + E_, 0, 3);   // global_prefetch for t+2
    }
    __syncthreads();   // x(t) staged everywhere; h(t-1) writes (buf prv) done

    // capture A fragments (shared across the 4 gate tiles)
    const __bf16* xr = lds_x + cur * XBUF;
    const __bf16* hr = lds_h + prv * HBUF;
    v16bf Ax, Ah[4];
    #pragma unroll
    for (int e = 0; e < 8; ++e) {
      Ax[e]     = xr[arow * XS + klo + e];
      Ax[e + 8] = xr[arow * XS + klo + 16 + e];
    }
    #pragma unroll
    for (int kc = 0; kc < 4; ++kc) {
      #pragma unroll
      for (int e = 0; e < 8; ++e) {
        Ah[kc][e]     = hr[arow * HS + kc * 32 + klo + e];
        Ah[kc][e + 8] = hr[arow * HS + kc * 32 + klo + 16 + e];
      }
    }

    // gates = bias + x@W_ih^T + h@W_hh^T : 4 independent 5-deep WMMA chains.
    // First WMMA reads persistent biasv as C (src2) while writing acc (dst).
    v8f acc[4];
    #pragma unroll
    for (int q = 0; q < 4; ++q) {
      acc[q] = __builtin_amdgcn_wmma_f32_16x16x32_bf16(false, Ax, false, Bx[q],
                                                       (short)0, biasv[q], false, false);
      #pragma unroll
      for (int kc = 0; kc < 4; ++kc)
        acc[q] = __builtin_amdgcn_wmma_f32_16x16x32_bf16(false, Ah[kc], false, Bh[q][kc],
                                                         (short)0, acc[q], false, false);
    }

    // LSTM cell directly on accumulators (torch gate order i,f,g,o)
    __bf16* hw = lds_h + cur * HBUF;
    #pragma unroll
    for (int r = 0; r < 8; ++r) {
      const float i_ = hsigmoid(acc[0][r]);
      const float f_ = hsigmoid(acc[1][r]);
      const float g_ = htanh(acc[2][r]);
      const float o_ = hsigmoid(acc[3][r]);
      const float cn = __builtin_fmaf(f_, c_st[r], i_ * g_);
      const float hn = o_ * htanh(cn);
      const bool  on = (t < len8[r]);
      c_st[r] = on ? cn : c_st[r];
      h_st[r] = on ? hn : h_st[r];
      const int m = r + hiHalf * 8;
      outs[((size_t)(b0 + m) * T_ + t) * H_ + j] = on ? hn : 0.f;
      hw[m * HS + j] = (__bf16)h_st[r];    // h(t) -> buffer cur, read at t+1
    }
    // single barrier per step: next iteration's __syncthreads orders all of
    // this step's lds writes; double buffering removes the WAR hazards.
  }

  // ---- final h, c ----
  #pragma unroll
  for (int r = 0; r < 8; ++r) {
    const int m = r + hiHalf * 8;
    h_out[(b0 + m) * H_ + j] = h_st[r];
    c_out[(b0 + m) * H_ + j] = c_st[r];
  }
}

// ---------------------------------------------------------------------------
extern "C" void kernel_launch(void* const* d_in, const int* in_sizes, int n_in,
                              void* d_out, int out_size, void* d_ws, size_t ws_size,
                              hipStream_t stream) {
  (void)in_sizes; (void)n_in; (void)out_size; (void)d_ws; (void)ws_size;
  const int*   obs_ids       = (const int*)d_in[0];
  const int*   obs_slot      = (const int*)d_in[1];
  const int*   action_ids    = (const int*)d_in[2];
  const int*   is_action     = (const int*)d_in[3];
  const int*   input_lengths = (const int*)d_in[4];
  const float* action_emb    = (const float*)d_in[5];
  const float* obs_emb       = (const float*)d_in[6];
  const float* W_ih          = (const float*)d_in[7];
  const float* W_hh          = (const float*)d_in[8];
  const float* b_ih          = (const float*)d_in[9];
  const float* b_hh          = (const float*)d_in[10];

  float* out      = (float*)d_out;
  float* outputs  = out;                               // [B,T,H]
  float* h_out    = out + (size_t)B_ * T_ * H_;        // [1,B,H]
  float* c_out    = h_out + (size_t)B_ * H_;           // [1,B,H]
  float* embedded = c_out + (size_t)B_ * H_;           // [B,T,E]

  embed_kernel<<<(B_ * T_) / 8, 256, 0, stream>>>(
      obs_ids, obs_slot, action_ids, is_action, action_emb, obs_emb, embedded);

  lstm_kernel<<<B_ / 16, 256, 0, stream>>>(
      embedded, input_lengths, W_ih, W_hh, b_ih, b_hh, outputs, h_out, c_out);
}
